// YMLocal_1434519076945
// MI455X (gfx1250) — matrix operand order repaired and checked by reading, not compiled, not run on hardware
//
#include <hip/hip_runtime.h>

#define NPIX 9216
#define CCH  256
#define KC   32
// sqrt( 32^-0.5 * log2(e) ): folded into stored Q so S^T emerges in exp2 domain
#define QSCALE 0.50501001f
#define NB   2

typedef __attribute__((ext_vector_type(16))) __bf16 v16bf;
typedef __attribute__((ext_vector_type(8)))  float  v8f;

union FragB16 {
  v16bf  v;
  uint4  q[2];
  __bf16 e[16];
};

// ---------------------------------------------------------------------------
// Kernel 1: per-pixel projections  q = (Wq*human + bq)*QSCALE  (bf16 [B][N][32])
//                                  v =  Wd*x + bd              (bf16 [B][32][N])
// ---------------------------------------------------------------------------
__global__ __launch_bounds__(256) void proj_kernel(
    const float* __restrict__ x, const float* __restrict__ human,
    const float* __restrict__ Wq, const float* __restrict__ bq,
    const float* __restrict__ Wd, const float* __restrict__ bd,
    __bf16* __restrict__ Q, __bf16* __restrict__ Vt) {
  __shared__ float sH[CCH][32];
  __shared__ float sX[CCH][32];
  const int b  = blockIdx.x / (NPIX / 32);
  const int n0 = (blockIdx.x % (NPIX / 32)) * 32;
  const int t  = threadIdx.x;
  for (int idx = t; idx < CCH * 32; idx += 256) {
    const int c = idx >> 5, px = idx & 31;
    const size_t g = ((size_t)b * CCH + c) * NPIX + n0 + px;
    sH[c][px] = human[g];
    sX[c][px] = x[g];
  }
  __syncthreads();
  const int px = t & 31;
  const int kb = t >> 5;  // uniform per wave
#pragma unroll
  for (int i = 0; i < 4; ++i) {
    const int k = kb + 8 * i;
    float aq = bq[k], av = bd[k];
    const float* wq = Wq + k * CCH;
    const float* wd = Wd + k * CCH;
    for (int c = 0; c < CCH; ++c) {
      aq = fmaf(wq[c], sH[c][px], aq);
      av = fmaf(wd[c], sX[c][px], av);
    }
    Q[((size_t)b * NPIX + n0 + px) * KC + k]  = (__bf16)(aq * QSCALE);
    Vt[((size_t)b * KC + k) * NPIX + n0 + px] = (__bf16)av;
  }
}

// ---------------------------------------------------------------------------
// Fragment loaders (row-major bf16 [row][32], 16B-aligned contiguous loads)
// A-frag (16-bit 16x32): lane l -> row l%16; K = {0..7,16..23} (l<16) else
// {8..15,24..31}.  B-frag: lane l -> col l%16; K same as A-frag.
// ---------------------------------------------------------------------------
__device__ __forceinline__ v16bf load_fragA(const __bf16* base, int row0, int lane) {
  const int r = row0 + (lane & 15);
  const int h = lane >> 4;
  const __bf16* p = base + (size_t)r * KC;
  FragB16 f;
  f.q[0] = *reinterpret_cast<const uint4*>(p + 8 * h);
  f.q[1] = *reinterpret_cast<const uint4*>(p + 8 * h + 16);
  return f.v;
}
__device__ __forceinline__ v16bf load_fragB_Q(const __bf16* base, int col0, int lane) {
  const int r = col0 + (lane & 15);
  const int h = lane >> 4;
  const __bf16* p = base + (size_t)r * KC + 8 * h;
  FragB16 f;
  f.q[0] = *reinterpret_cast<const uint4*>(p);
  f.q[1] = *reinterpret_cast<const uint4*>(p + 16);
  return f.v;
}
__device__ __forceinline__ v16bf load_fragB_V(const __bf16* VtB, int ch, int m0, int lane) {
  const int h = lane >> 4;
  const __bf16* p = VtB + (size_t)ch * NPIX + m0 + 8 * h;
  FragB16 f;
  f.q[0] = *reinterpret_cast<const uint4*>(p);
  f.q[1] = *reinterpret_cast<const uint4*>(p + 16);
  return f.v;
}

// One 16-query-row tile step over 32 keys: 2 S^T WMMAs + online softmax
// (exp2 domain, lazy rescale) + 2 P*V WMMAs.
__device__ __forceinline__ void attn_step(
    const v16bf ka0, const v16bf ka1, const v16bf qf,
    const v16bf vb0, const v16bf vb1, const int hb,
    float& rmax, float& rsum, v8f& acc0, v8f& acc1) {
  const v8f zero = {0.f, 0.f, 0.f, 0.f, 0.f, 0.f, 0.f, 0.f};
  // S^T tiles (already in exp2 domain): one query row per lane
  v8f st0 = __builtin_amdgcn_wmma_f32_16x16x32_bf16(
      false, ka0, false, qf, (short)0, zero, false, false);
  v8f st1 = __builtin_amdgcn_wmma_f32_16x16x32_bf16(
      false, ka1, false, qf, (short)0, zero, false, false);

  float tmax = -__builtin_inff();
#pragma unroll
  for (int j = 0; j < 8; ++j) tmax = fmaxf(tmax, fmaxf(st0[j], st1[j]));
  tmax = fmaxf(tmax, __shfl_xor(tmax, 16, 32));  // combine row halves

  const bool  grew = tmax > rmax;
  const float nmax = grew ? tmax : rmax;
  // wave-uniform branch: EXEC stays all-ones (WMMA-legal); skipped once the
  // running max stops growing.
  if (__builtin_amdgcn_ballot_w32(grew)) {
    const float alpha = __builtin_amdgcn_exp2f(rmax - nmax);
    rsum *= alpha;
#pragma unroll
    for (int j = 0; j < 8; ++j) {
      const float a = __shfl(alpha, j + hb, 32);  // alpha of row j+hb
      acc0[j] *= a;
      acc1[j] *= a;
    }
    rmax = nmax;
  }

  FragB16 pa;  // P packs lane-locally into the A-fragment layout (no shuffles)
  float psum = 0.f;
#pragma unroll
  for (int j = 0; j < 8; ++j) {
    const float p = __builtin_amdgcn_exp2f(st0[j] - nmax);
    psum += p;
    pa.e[j] = (__bf16)p;
  }
#pragma unroll
  for (int j = 0; j < 8; ++j) {
    const float p = __builtin_amdgcn_exp2f(st1[j] - nmax);
    psum += p;
    pa.e[8 + j] = (__bf16)p;
  }
  rsum += psum;

  acc0 = __builtin_amdgcn_wmma_f32_16x16x32_bf16(
      false, pa.v, false, vb0, (short)0, acc0, false, false);
  acc1 = __builtin_amdgcn_wmma_f32_16x16x32_bf16(
      false, pa.v, false, vb1, (short)0, acc1, false, false);
}

// ---------------------------------------------------------------------------
// Kernel 2: flash attention. One wave owns 32 query rows (2 tiles) -> each
// K/V fragment pair feeds 8 WMMAs, halving L2 traffic vs 16-row tiling.
// ---------------------------------------------------------------------------
__global__ __launch_bounds__(128) void attn_kernel(
    const __bf16* __restrict__ Q, const __bf16* __restrict__ Vt,
    float* __restrict__ Ctx) {
  const int lane = threadIdx.x & 31;
  const int wid  = blockIdx.x * (blockDim.x >> 5) + (threadIdx.x >> 5);
  const int b  = wid / (NPIX / 32);
  const int n0 = (wid % (NPIX / 32)) * 32;
  const __bf16* Qb  = Q  + (size_t)b * NPIX * KC;
  const __bf16* VtB = Vt + (size_t)b * KC * NPIX;
  const int hb = (lane >> 4) << 3;  // 0 or 8

  const v16bf qf0 = load_fragB_Q(Qb, n0, lane);
  const v16bf qf1 = load_fragB_Q(Qb, n0 + 16, lane);
  v8f acc00 = {0.f, 0.f, 0.f, 0.f, 0.f, 0.f, 0.f, 0.f};
  v8f acc01 = acc00, acc10 = acc00, acc11 = acc00;
  float rmax0 = -__builtin_inff(), rsum0 = 0.f;
  float rmax1 = -__builtin_inff(), rsum1 = 0.f;

  for (int m0 = 0; m0 < NPIX; m0 += 32) {
    const v16bf ka0 = load_fragA(Qb, m0, lane);
    const v16bf ka1 = load_fragA(Qb, m0 + 16, lane);
    const v16bf vb0 = load_fragB_V(VtB, lane & 15, m0, lane);
    const v16bf vb1 = load_fragB_V(VtB, 16 + (lane & 15), m0, lane);
    if (m0 + 32 < NPIX) {
      __builtin_prefetch(Qb + (size_t)(m0 + 32 + (lane & 15)) * KC, 0, 1);
      __builtin_prefetch(VtB + (size_t)(lane & 31) * NPIX + m0 + 32, 0, 1);
    }
    attn_step(ka0, ka1, qf0, vb0, vb1, hb, rmax0, rsum0, acc00, acc01);
    attn_step(ka0, ka1, qf1, vb0, vb1, hb, rmax1, rsum1, acc10, acc11);
  }

  float* ctxB = Ctx + (size_t)b * NPIX * KC;
  {
    const float ltot = rsum0 + __shfl_xor(rsum0, 16, 32);
    const float inv  = 1.0f / ltot;
#pragma unroll
    for (int j = 0; j < 8; ++j) {
      const float r = __shfl(inv, j + hb, 32);
      const size_t row = (size_t)(n0 + j + hb);
      ctxB[row * KC + (lane & 15)]      = acc00[j] * r;
      ctxB[row * KC + 16 + (lane & 15)] = acc01[j] * r;
    }
  }
  {
    const float ltot = rsum1 + __shfl_xor(rsum1, 16, 32);
    const float inv  = 1.0f / ltot;
#pragma unroll
    for (int j = 0; j < 8; ++j) {
      const float r = __shfl(inv, j + hb, 32);
      const size_t row = (size_t)(n0 + 16 + j + hb);
      ctxB[row * KC + (lane & 15)]      = acc10[j] * r;
      ctxB[row * KC + 16 + (lane & 15)] = acc11[j] * r;
    }
  }
}

// ---------------------------------------------------------------------------
// Kernel 3: y = relu(Wf*ctx + bf); out = x + Wu*y + bu   (fused, LDS-staged)
// ---------------------------------------------------------------------------
__global__ __launch_bounds__(256) void out_kernel(
    const float* __restrict__ x,
    const float* __restrict__ Wf, const float* __restrict__ bf,
    const float* __restrict__ Wu, const float* __restrict__ bu,
    const float* __restrict__ Ctx, float* __restrict__ out) {
  __shared__ float sWu[CCH][KC];
  __shared__ float sWf[KC][KC];
  __shared__ float sCtx[32][KC + 1];
  __shared__ float sY[32][KC + 1];
  __shared__ float sBu[CCH];
  __shared__ float sBf[KC];
  const int b  = blockIdx.x / (NPIX / 32);
  const int n0 = (blockIdx.x % (NPIX / 32)) * 32;
  const int t  = threadIdx.x;
  for (int idx = t; idx < CCH * KC; idx += 256) sWu[idx >> 5][idx & 31] = Wu[idx];
  for (int idx = t; idx < KC * KC; idx += 256)  sWf[idx >> 5][idx & 31] = Wf[idx];
  for (int idx = t; idx < 32 * KC; idx += 256) {
    const int px = idx >> 5, ch = idx & 31;
    sCtx[px][ch] = Ctx[((size_t)b * NPIX + n0 + px) * KC + ch];
  }
  if (t < CCH) sBu[t] = bu[t];
  if (t < KC)  sBf[t] = bf[t];
  __syncthreads();
  const int px = t & 31;
  const int g  = t >> 5;  // uniform per wave
#pragma unroll
  for (int i = 0; i < 4; ++i) {
    const int k = g + 8 * i;
    float s = sBf[k];
#pragma unroll
    for (int j = 0; j < KC; ++j) s = fmaf(sWf[k][j], sCtx[px][j], s);
    sY[px][k] = fmaxf(s, 0.f);
  }
  __syncthreads();
  for (int i = 0; i < 32; ++i) {
    const int c = g * 32 + i;
    float s = sBu[c];
#pragma unroll
    for (int j = 0; j < KC; ++j) s = fmaf(sWu[c][j], sY[px][j], s);
    const size_t gi = ((size_t)b * CCH + c) * NPIX + n0 + px;
    out[gi] = x[gi] + s;
  }
}

// ---------------------------------------------------------------------------
extern "C" void kernel_launch(void* const* d_in, const int* in_sizes, int n_in,
                              void* d_out, int out_size, void* d_ws, size_t ws_size,
                              hipStream_t stream) {
  const float* x     = (const float*)d_in[0];
  const float* human = (const float*)d_in[1];
  const float* Wq    = (const float*)d_in[2];
  const float* bq    = (const float*)d_in[3];
  const float* Wd    = (const float*)d_in[4];
  const float* bd    = (const float*)d_in[5];
  const float* Wf    = (const float*)d_in[6];
  const float* bf    = (const float*)d_in[7];
  const float* Wu    = (const float*)d_in[8];
  const float* bu    = (const float*)d_in[9];
  float* out = (float*)d_out;

  char* ws = (char*)d_ws;
  const size_t qBytes = (size_t)NB * NPIX * KC * sizeof(__bf16);  // 1.125 MB
  __bf16* Q   = (__bf16*)ws;
  __bf16* Vt  = (__bf16*)(ws + qBytes);
  float*  Ctx = (float*)(ws + 2 * qBytes);  // NB*NPIX*KC fp32

  proj_kernel<<<NB * (NPIX / 32), 256, 0, stream>>>(x, human, Wq, bq, Wd, bd, Q, Vt);
  attn_kernel<<<(NB * (NPIX / 32)) / 4, 128, 0, stream>>>(Q, Vt, Ctx);
  out_kernel<<<NB * (NPIX / 32), 256, 0, stream>>>(x, Wf, bf, Wu, bu, Ctx, out);
}
